// VQCAMHead_3556232921211
// MI455X (gfx1250) — matrix operand order repaired and checked by reading, not compiled, not run on hardware
//
#include <hip/hip_runtime.h>
#include <hip/hip_bf16.h>

// MI455X / gfx1250, wave32. Fused VQ head:
//   prep:     centers f32 -> bf16 (workspace) + wc[n] = dot(w, centers[n])
//   main:     GEMM (x^T · centers^T) via v_wmma_f32_16x16x32_bf16, fused with
//             online argmax / sum-exp over the codebook axis; writes heatmap.
//   finalize: deterministic mean over H*W per batch -> logits.
//
// Streaming operands (x: 134MB, gumbel: 268MB) use non-temporal loads so the
// hot bf16 codebook (1MB, broadcast to all 512 WGs) stays L2-resident.
// Centers staging uses GLOBAL_LOAD_ASYNC_TO_LDS_B128 (ASYNCcnt path) —
// builtin confirmed present by the previous round's diagnostic.

typedef __bf16 bf16_t;
typedef __attribute__((ext_vector_type(16))) __bf16        v16bf;
typedef __attribute__((ext_vector_type(8)))  float         v8f;
typedef __attribute__((ext_vector_type(4)))  unsigned int  u32x4;
typedef __attribute__((ext_vector_type(4)))  float         f32x4;

// Builtin's parameter type per hipcc diagnostic:
//   "__attribute__((__vector_size__(4 * sizeof(int)))) int __device__ *"
typedef int v4i __attribute__((vector_size(16)));
typedef __attribute__((address_space(1))) v4i* gv4i_p;   // global (__device__)
typedef __attribute__((address_space(3))) v4i* lv4i_p;   // LDS (__shared__)

#define B_   16
#define C_   512
#define HW_  4096
#define N_   1024
#define INV_TAU 0.1f

#if defined(__has_builtin)
#  if __has_builtin(__builtin_amdgcn_global_load_async_to_lds_b128)
#    define HAVE_ASYNC_LDS 1
#  endif
#endif
#ifndef HAVE_ASYNC_LDS
#  define HAVE_ASYNC_LDS 0
#endif

__device__ __forceinline__ void wait_asynccnt0() {
#if defined(__has_builtin) && __has_builtin(__builtin_amdgcn_s_wait_asynccnt)
    __builtin_amdgcn_s_wait_asynccnt(0);
#else
    asm volatile("s_wait_asynccnt 0x0" ::: "memory");
#endif
}

// ---------------- prep: centers -> bf16, wc[n] = dot(w, centers[n]) ----------
__global__ __launch_bounds__(256) void vq_prep(const float* __restrict__ centers,
                                               const float* __restrict__ w,
                                               bf16_t* __restrict__ cbf,
                                               float* __restrict__ wc)
{
    int lane = threadIdx.x & 31;
    int wave = threadIdx.x >> 5;
    int n    = blockIdx.x * 8 + wave;          // one center row per wave
    const float* crow = centers + (size_t)n * C_;

    union { bf16_t h[16]; u32x4 q[2]; } u;
    float dot = 0.f;
#pragma unroll
    for (int i = 0; i < 16; ++i) {
        float f = crow[lane * 16 + i];
        dot += f * w[lane * 16 + i];
        u.h[i] = (bf16_t)f;                    // RNE f32 -> bf16
    }
    u32x4* dst = (u32x4*)(cbf + (size_t)n * C_ + lane * 16);
    dst[0] = u.q[0]; dst[1] = u.q[1];

#pragma unroll
    for (int m = 16; m >= 1; m >>= 1) dot += __shfl_xor(dot, m);
    if (lane == 0) wc[n] = dot;
}

// ---------------- main fused kernel -----------------------------------------
__global__ __launch_bounds__(256) void vq_main(const float* __restrict__ x,
                                               const float* __restrict__ gumbel,
                                               const bf16_t* __restrict__ cbf,
                                               const float* __restrict__ wc,
                                               const float* __restrict__ bias,
                                               float* __restrict__ heat_out)
{
    // 32 center rows, each padded 1024B -> 1040B so 16-lane fragment reads
    // (row stride 260 dwords == 4 mod 64 banks) are bank-conflict free.
    __shared__ __align__(16) unsigned char ldsB[32 * 1040];

    const int tid   = threadIdx.x;
    const int lane  = tid & 31;
    const int wave  = tid >> 5;
    const int mlane = lane & 15;
    const int khalf = lane >> 4;

    const int p0   = blockIdx.x * 128 + wave * 16;   // 16 pixels per wave
    const int bidx = p0 >> 12;                        // batch (tile never crosses b)
    const int hw0  = p0 & 4095;

    // ---- A tile (16 pixels x 512 ch) -> bf16 register fragments ----
    // ISA 16-bit A 16x32 layout: lane m=L&15, khalf=L>>4;
    // element i: K = kb*32 + (i<8 ? khalf*8+i : 16 + khalf*8 + (i-8)).
    const float* xp = x + (size_t)bidx * ((size_t)C_ * HW_) + hw0 + mlane;
    v16bf a[16];
#pragma unroll
    for (int kb = 0; kb < 16; ++kb) {
#pragma unroll
        for (int half = 0; half < 2; ++half) {
            int cb = kb * 32 + half * 16 + khalf * 8;
#pragma unroll
            for (int e = 0; e < 8; ++e) {
                float f = __builtin_nontemporal_load(&xp[(size_t)(cb + e) * HW_]);
                a[kb][half * 8 + e] = (bf16_t)f;
            }
        }
    }

    // per-lane running softmax state; slot r <-> pixel m = r + 8*khalf,
    // this lane covers codebook entries n with (n % 16) == mlane.
    float smax[8], ssum[8]; int sidx[8];
#pragma unroll
    for (int r = 0; r < 8; ++r) { smax[r] = -3.4e38f; ssum[r] = 0.f; sidx[r] = 0; }

    const size_t gbase = (size_t)bidx * ((size_t)N_ * HW_) + hw0 + khalf * 8;

    for (int c0 = 0; c0 < N_; c0 += 32) {
        // cooperative stage: 32 bf16 center rows -> LDS (2048 x 16B)
#if HAVE_ASYNC_LDS
#pragma unroll
        for (int i = 0; i < 8; ++i) {
            int f   = tid + i * 256;          // 16B-unit index, 2048 total
            int row = f >> 6, col = f & 63;   // 64 units per row
            __builtin_amdgcn_global_load_async_to_lds_b128(
                (gv4i_p)(cbf + ((size_t)(c0 + row)) * C_ + col * 8),
                (lv4i_p)(&ldsB[row * 1040 + col * 16]),
                0, 0);
        }
        wait_asynccnt0();
        __syncthreads();
#else
#pragma unroll
        for (int i = 0; i < 8; ++i) {
            int f   = tid + i * 256;          // 16B-unit index, 2048 total
            int row = f >> 6, col = f & 63;   // 64 units per row
            *(u32x4*)(&ldsB[row * 1040 + col * 16]) =
                *(const u32x4*)(cbf + ((size_t)(c0 + row)) * C_ + col * 8);
        }
        __syncthreads();
#endif

#pragma unroll
        for (int nb = 0; nb < 2; ++nb) {
            const int nloc = nb * 16 + mlane;
            v8f acc = {0.f,0.f,0.f,0.f,0.f,0.f,0.f,0.f};
            // B 32x16 fragment: lane n = L&15, K = kb*32 + (i<8 ? khalf*8+i : 16 + khalf*8 + (i-8))
#pragma unroll
            for (int kb = 0; kb < 16; ++kb) {
                union { u32x4 q[2]; v16bf v; } ub;
                const u32x4* bp =
                    (const u32x4*)(&ldsB[nloc * 1040 + (kb * 32 + khalf * 8) * 2]);
                ub.q[0] = bp[0]; ub.q[1] = bp[2];
                acc = __builtin_amdgcn_wmma_f32_16x16x32_bf16(
                        false, a[kb], false, ub.v, (short)0, acc, false, false);
            }
            // epilogue: gumbel + online max/argmax/sum-exp (no cross-lane ops)
            const int ng = c0 + nloc;
            const float* gp = gumbel + gbase + (size_t)ng * HW_;
            f32x4 g0 = __builtin_nontemporal_load((const f32x4*)gp);
            f32x4 g1 = __builtin_nontemporal_load((const f32x4*)(gp + 4));
            float g[8] = {g0.x,g0.y,g0.z,g0.w,g1.x,g1.y,g1.z,g1.w};
#pragma unroll
            for (int r = 0; r < 8; ++r) {
                float l = acc[r] + g[r];
                ssum[r] += __expf(l * INV_TAU);       // no max-shift: |l|/tau small
                if (l > smax[r]) { smax[r] = l; sidx[r] = ng; }
            }
        }
        __syncthreads();
    }

    // ---- final cross-lane combine over the 16-lane n-groups + store --------
    const float bv = bias[0];
#pragma unroll
    for (int r = 0; r < 8; ++r) {
        float mx = smax[r], sm = ssum[r]; int ix = sidx[r];
#pragma unroll
        for (int m = 1; m <= 8; m <<= 1) {
            float omx = __shfl_xor(mx, m);
            float osm = __shfl_xor(sm, m);
            int   oix = __shfl_xor(ix, m);
            sm += osm;
            if (omx > mx) { mx = omx; ix = oix; }
        }
        if (mlane == 0) {
            float s     = __expf(mx * INV_TAU) / sm;  // softmax value at argmax
            float scale = (1.0f - s) + s;             // straight-through residue
            heat_out[p0 + khalf * 8 + r] = scale * wc[ix] + bv;
        }
    }
}

// ---------------- logits = mean over H*W (deterministic, no atomics) --------
__global__ __launch_bounds__(256) void vq_finalize(const float* __restrict__ heat,
                                                   float* __restrict__ logits)
{
    __shared__ float red[8];
    int tid = threadIdx.x, lane = tid & 31, wave = tid >> 5;
    const float* hp = heat + (size_t)blockIdx.x * HW_;
    float s = 0.f;
    for (int i = tid; i < HW_; i += 256) s += hp[i];
#pragma unroll
    for (int m = 16; m >= 1; m >>= 1) s += __shfl_xor(s, m);
    if (lane == 0) red[wave] = s;
    __syncthreads();
    if (tid == 0) {
        float t = 0.f;
#pragma unroll
        for (int i = 0; i < 8; ++i) t += red[i];
        logits[blockIdx.x] = t * (1.0f / HW_);
    }
}

// ---------------- launch ----------------------------------------------------
extern "C" void kernel_launch(void* const* d_in, const int* in_sizes, int n_in,
                              void* d_out, int out_size, void* d_ws, size_t ws_size,
                              hipStream_t stream) {
    const float* x       = (const float*)d_in[0];   // [16,512,64,64]
    const float* centers = (const float*)d_in[1];   // [1024,512]
    const float* w       = (const float*)d_in[2];   // [1,512]
    const float* b       = (const float*)d_in[3];   // [1]
    const float* gumbel  = (const float*)d_in[4];   // [16,1024,64,64]
    float* out = (float*)d_out;                     // 65536 heatmap + 16 logits

    bf16_t* cbf = (bf16_t*)d_ws;                                   // 1 MB
    float*  wc  = (float*)((char*)d_ws + (size_t)N_ * C_ * sizeof(bf16_t)); // 4 KB

    vq_prep    <<<N_ / 8,          256, 0, stream>>>(centers, w, cbf, wc);
    vq_main    <<<(B_*HW_) / 128,  256, 0, stream>>>(x, gumbel, cbf, wc, b, out);
    vq_finalize<<<B_,              256, 0, stream>>>(out, out + (size_t)B_ * HW_);
}
